// MiMoV2FlashAttention_79259326480648
// MI455X (gfx1250) — matrix operand order repaired and checked by reading, not compiled, not run on hardware
//
#include <hip/hip_runtime.h>
#include <cstdint>

// ---- problem constants (match reference) ----
constexpr int BB   = 2;
constexpr int SS   = 2048;
constexpr int HIDN = 2048;
constexpr int NH   = 16;     // query heads
constexpr int NKV  = 4;      // kv heads
constexpr int DH   = 128;    // head dim
constexpr int DV   = 128;    // value head dim
constexpr int RP   = 64;     // rope dims
constexpr int NREP = NH / NKV;
constexpr float SCALE = 0.08838834764831845f;   // 128^-0.5
constexpr float NEGF  = -1000000000.0f;

typedef __bf16 bf16;
typedef __attribute__((ext_vector_type(16))) __bf16 v16bf;
typedef __attribute__((ext_vector_type(8)))  float  v8f;
typedef unsigned int v4u __attribute__((ext_vector_type(4)));
typedef int v8i __attribute__((ext_vector_type(8)));
typedef int v4i __attribute__((ext_vector_type(4)));

__device__ inline v8f v8f_zero() {
  v8f z;
#pragma unroll
  for (int i = 0; i < 8; ++i) z[i] = 0.0f;
  return z;
}

__device__ inline bf16 to_bf16(float x) { return (bf16)x; }
__device__ inline bf16 to_bf16(bf16 x)  { return x; }

template <typename T> __device__ inline T from_f32(float x);
template <> __device__ inline float from_f32<float>(float x) { return x; }
template <> __device__ inline bf16  from_f32<bf16>(float x)  { return (bf16)x; }

// =====================================================================
// Tensor Data Mover: DMA a rows x cols f32 tile (row stride = stride
// elems) from global into LDS at lds_byte_off. LDS layout gets 1 dword
// of padding after every 64 dwords (pad_interval=5, pad_amount=0), so a
// 64-wide f32 tile lands with a 65-dword row pitch (bank-conflict free
// column reads). Tracked by TENSORcnt.
// =====================================================================
__device__ inline void tdm_load_tile_f32(uint32_t lds_byte_off,
                                         const float* gptr,
                                         int rows, int cols,
                                         int stride_elems,
                                         int tensor_rows, int tensor_cols)
{
  unsigned long long ga = (unsigned long long)(uintptr_t)gptr;

  v4u g0;
  g0[0] = 1u;                                            // count=1
  g0[1] = lds_byte_off;                                  // lds_addr
  g0[2] = (unsigned int)(ga & 0xFFFFFFFFu);              // global_addr lo
  g0[3] = (unsigned int)((ga >> 32) & 0x1FFFFFFu)        // global_addr hi
        | (2u << 30);                                    // type=2 (image)

  unsigned int d0 = (unsigned int)tensor_cols;
  unsigned int d1 = (unsigned int)tensor_rows;
  unsigned long long s0 = (unsigned long long)stride_elems;

  v8i g1;
  g1[0] = (int)((2u << 16)        // data_size = 4B
              | (1u << 20)        // pad_enable
              | (5u << 22)        // pad_interval: 64 dwords
              | (0u << 25));      // pad_amount: 1 dword
  g1[1] = (int)((d0 & 0xFFFFu) << 16);                          // dim0 lo
  g1[2] = (int)((d0 >> 16) | ((d1 & 0xFFFFu) << 16));           // dim0 hi | dim1 lo
  g1[3] = (int)((d1 >> 16) | ((unsigned int)cols << 16));       // dim1 hi | tile_dim0
  g1[4] = (int)((unsigned int)rows & 0xFFFFu);                  // tile_dim1 (tile_dim2=0)
  g1[5] = (int)(unsigned int)(s0 & 0xFFFFFFFFu);                // dim0_stride lo
  g1[6] = (int)(unsigned int)((s0 >> 32) & 0xFFFFu);            // dim0_stride hi
  g1[7] = 0;

  v4i z4 = {0, 0, 0, 0};
#if defined(__clang_major__) && (__clang_major__ >= 23)
  v8i z8 = {0, 0, 0, 0, 0, 0, 0, 0};
  __builtin_amdgcn_tensor_load_to_lds(g0, g1, z4, z4, z8, 0);
#else
  __builtin_amdgcn_tensor_load_to_lds(g0, g1, z4, z4, 0);
#endif
}

// =====================================================================
// GEMM: Y[M,N] = X[M,K] @ W[K,N]   (bf16 WMMA, f32 accumulate)
// block = 256 threads = 8 waves; each wave computes a 16x64 tile.
// W tiles (32x64 f32) are staged to LDS by the TDM, double buffered.
// =====================================================================
constexpr int WPITCH = 65;   // dwords per staged row (64 data + 1 TDM pad)

template <typename TIN, typename TOUT>
__global__ __launch_bounds__(256) void gemm_bf16_wmma(
    const TIN* __restrict__ X, const float* __restrict__ W,
    TOUT* __restrict__ Y, int M, int N, int K)
{
  __shared__ float Ws[2][32 * WPITCH];

  const int lane = threadIdx.x & 31;
  const int wave = threadIdx.x >> 5;
  const int n16  = lane & 15;
  const int h2   = lane >> 4;

  const int row0 = blockIdx.y * 128 + wave * 16;
  const int col0 = blockIdx.x * 64;
  const int arow = row0 + n16;

  v8f acc[4];
#pragma unroll
  for (int t = 0; t < 4; ++t) acc[t] = v8f_zero();

  const int nk = K / 32;

  if (threadIdx.x == 0)   // wave 0 drives the TDM (EXEC ignored by tensor ops)
    tdm_load_tile_f32((uint32_t)(uintptr_t)&Ws[0][0], W + col0,
                      32, 64, N, K, N);

  for (int i = 0; i < nk; ++i) {
    if (threadIdx.x == 0) {
      if (i + 1 < nk) {
        tdm_load_tile_f32((uint32_t)(uintptr_t)&Ws[(i + 1) & 1][0],
                          W + (size_t)(i + 1) * 32 * N + col0,
                          32, 64, N, K, N);
        __builtin_amdgcn_s_wait_tensorcnt((short)1);   // tile i landed
      } else {
        __builtin_amdgcn_s_wait_tensorcnt((short)0);
      }
    }
    __syncthreads();

    const float* wt = Ws[i & 1];
    const int k0 = i * 32;

    // A fragment: 16x32 tile from global, row = lane&15
    v16bf a;
#pragma unroll
    for (int j = 0; j < 16; ++j) {
      int kk = k0 + (j < 8 ? h2 * 8 + j : 16 + h2 * 8 + (j - 8));
      a[j] = to_bf16(X[(size_t)arow * K + kk]);
    }
#pragma unroll
    for (int t = 0; t < 4; ++t) {
      // B fragment from LDS: col = lane&15, contraction = table order (h2*8 + j / 16 + h2*8 + j-8)
      v16bf bfr;
#pragma unroll
      for (int j = 0; j < 16; ++j)
        bfr[j] = (bf16)wt[(j < 8 ? h2 * 8 + j : 16 + h2 * 8 + (j - 8)) * WPITCH + t * 16 + n16];
      acc[t] = __builtin_amdgcn_wmma_f32_16x16x32_bf16(
          false, a, false, bfr, (short)0, acc[t], false, false);
    }
    __syncthreads();   // buffer (i&1) free for reuse at step i+2
  }

#pragma unroll
  for (int t = 0; t < 4; ++t)
#pragma unroll
    for (int r = 0; r < 8; ++r) {
      int m = row0 + r + 8 * h2;
      int n = col0 + t * 16 + n16;
      Y[(size_t)m * N + n] = from_f32<TOUT>(acc[t][r]);
    }
}

// =====================================================================
// RoPE on bf16 Q and K (first RP dims); one thread per rotated pair.
// =====================================================================
__global__ void rope_kernel(bf16* __restrict__ Qb, bf16* __restrict__ Kb,
                            const float* __restrict__ cosb,
                            const float* __restrict__ sinb)
{
  int idx = blockIdx.x * blockDim.x + threadIdx.x;
  const int total = BB * SS * (NH + NKV) * (RP / 2);
  if (idx >= total) return;
  int d    = idx & (RP / 2 - 1);           // 0..31
  int rest = idx >> 5;
  int head = rest % (NH + NKV);
  int tok  = rest / (NH + NKV);            // b*SS + s

  float c1 = cosb[(size_t)tok * RP + d];
  float s1 = sinb[(size_t)tok * RP + d];
  float c2 = cosb[(size_t)tok * RP + d + RP / 2];
  float s2 = sinb[(size_t)tok * RP + d + RP / 2];

  bf16* p = (head < NH) ? (Qb + ((size_t)tok * NH + head) * DH)
                        : (Kb + ((size_t)tok * NKV + (head - NH)) * DH);
  float x1 = (float)p[d];
  float x2 = (float)p[d + RP / 2];
  p[d]          = (bf16)(x1 * c1 - x2 * s1);
  p[d + RP / 2] = (bf16)(x2 * c2 + x1 * s2);
}

// =====================================================================
// V transpose: Vb[(b*S+s)*NKV+kv][d] -> Vt[((b*NKV+kv)*DV+d)][s]
// makes attention B-fragment loads contiguous per lane.
// =====================================================================
__global__ void transpose_v(const bf16* __restrict__ Vb, bf16* __restrict__ Vt)
{
  int idx = blockIdx.x * blockDim.x + threadIdx.x;
  const int total = BB * SS * NKV * DV;
  if (idx >= total) return;
  int d  = idx & (DV - 1);
  int r  = idx >> 7;
  int kv = r % NKV;
  int t  = r / NKV;              // b*SS + s
  int b  = t / SS;
  int s  = t - b * SS;
  Vt[(((size_t)(b * NKV + kv)) * DV + d) * SS + s] = Vb[idx];
}

// =====================================================================
// zero-fill (probs upper triangle must be exactly 0)
// =====================================================================
__global__ void zero_f32(float* __restrict__ p, size_t n)
{
  size_t stride = (size_t)gridDim.x * blockDim.x;
  for (size_t i = (size_t)blockIdx.x * blockDim.x + threadIdx.x; i < n; i += stride)
    p[i] = 0.0f;
}

// =====================================================================
// Flash attention with sink logit + full probs output.
// 1 wave per (b, head, 16-row q tile). Two passes over causal key tiles.
// =====================================================================
__global__ __launch_bounds__(32) void attn_kernel(
    const bf16* __restrict__ Qb, const bf16* __restrict__ Kb,
    const bf16* __restrict__ Vt, const float* __restrict__ sink,
    float* __restrict__ probs, bf16* __restrict__ AOb)
{
  __shared__ bf16 Pl[16 * 32];   // P tile staged C-layout -> A-layout

  const int lane = threadIdx.x & 31;
  const int n16  = lane & 15;
  const int h2   = lane >> 4;
  const int qt   = blockIdx.x;   // q tile (16 rows)
  const int qh   = blockIdx.y;   // query head
  const int b    = blockIdx.z;
  const int kvh  = qh / NREP;

  // ---- load Q fragments (16 x 128 = 4 A-fragments of 16x32) ----
  const int qrowA = qt * 16 + n16;
  const bf16* qp = Qb + ((size_t)(b * SS + qrowA) * NH + qh) * DH;
  v16bf qf[4];
#pragma unroll
  for (int f = 0; f < 4; ++f)
#pragma unroll
    for (int j = 0; j < 16; ++j) {
      int kk = f * 32 + (j < 8 ? h2 * 8 + j : 16 + h2 * 8 + (j - 8));
      qf[f][j] = qp[kk];
    }

  // score tile for key tile kt -> per-lane sv[8] (rows r+8*h2, col n16)
  auto score = [&](int kt, float sv[8]) {
    v8f sc = v8f_zero();
    const bf16* kp = Kb + ((size_t)(b * SS + kt * 16 + n16) * NKV + kvh) * DH;
#pragma unroll
    for (int f = 0; f < 4; ++f) {
      v16bf kf;   // B-frag: N = key (lane&15), contraction = head dim
#pragma unroll
      for (int j = 0; j < 16; ++j) kf[j] = kp[f * 32 + (j < 8 ? h2 * 8 + j : 16 + h2 * 8 + (j - 8))];
      sc = __builtin_amdgcn_wmma_f32_16x16x32_bf16(
          false, qf[f], false, kf, (short)0, sc, false, false);
    }
#pragma unroll
    for (int r = 0; r < 8; ++r) {
      int qrow = qt * 16 + r + 8 * h2;
      int key  = kt * 16 + n16;
      sv[r] = sc[r] * SCALE + ((key > qrow) ? NEGF : 0.0f);
    }
  };

  // ---- pass 1: online max/sum, seeded by sink logit ----
  float m[8], l[8];
  const float snk = sink[qh];
#pragma unroll
  for (int r = 0; r < 8; ++r) { m[r] = snk; l[r] = 1.0f; }

  for (int kt = 0; kt <= qt; ++kt) {
    float sv[8];
    score(kt, sv);
#pragma unroll
    for (int r = 0; r < 8; ++r) {
      float rmax = sv[r];
      for (int off = 1; off < 16; off <<= 1)
        rmax = fmaxf(rmax, __shfl_xor(rmax, off, 32));
      float mn = fmaxf(m[r], rmax);
      float e  = __expf(sv[r] - mn);
      float rsum = e;
      for (int off = 1; off < 16; off <<= 1)
        rsum += __shfl_xor(rsum, off, 32);
      l[r] = l[r] * __expf(m[r] - mn) + rsum;
      m[r] = mn;
    }
  }
  float inv[8];
#pragma unroll
  for (int r = 0; r < 8; ++r) inv[r] = 1.0f / l[r];

  // ---- pass 2: write probs, accumulate O = P @ V ----
  v8f oacc[8];
#pragma unroll
  for (int t = 0; t < 8; ++t) oacc[t] = v8f_zero();

  const bf16* vp = Vt + ((size_t)(b * NKV + kvh)) * DV * SS;

  for (int kt = 0; kt <= qt; ++kt) {
    float sv[8];
    score(kt, sv);

    if ((kt & 1) == 0) {   // fresh 16x32 P buffer (zero-pad odd tail)
#pragma unroll
      for (int j = 0; j < 16; ++j) Pl[lane * 16 + j] = (bf16)0.0f;
      __syncthreads();
    }

#pragma unroll
    for (int r = 0; r < 8; ++r) {
      int qrow = qt * 16 + r + 8 * h2;
      int key  = kt * 16 + n16;
      float p  = __expf(sv[r] - m[r]) * inv[r];
      probs[(((size_t)(b * NH + qh)) * SS + qrow) * SS + key] = p;
      Pl[(r + 8 * h2) * 32 + (kt & 1) * 16 + n16] = (bf16)p;
    }
    __syncthreads();

    if ((kt & 1) == 1 || kt == qt) {
      const int kbase = (kt & ~1) * 16;
      // reload P as A-fragment (16x32)
      v16bf pa;
#pragma unroll
      for (int j = 0; j < 16; ++j) {
        int kk = (j < 8 ? h2 * 8 + j : 16 + h2 * 8 + (j - 8));
        pa[j] = Pl[n16 * 32 + kk];
      }
#pragma unroll
      for (int t = 0; t < 8; ++t) {
        v16bf vf;   // B-frag: contraction = key, N = value dim (contiguous)
#pragma unroll
        for (int j = 0; j < 16; ++j)
          vf[j] = vp[(size_t)(t * 16 + n16) * SS + kbase + (j < 8 ? h2 * 8 + j : 16 + h2 * 8 + (j - 8))];
        oacc[t] = __builtin_amdgcn_wmma_f32_16x16x32_bf16(
            false, pa, false, vf, (short)0, oacc[t], false, false);
      }
      __syncthreads();
    }
  }

  // ---- store attention output (already normalized), layout (B,S,H*DV) ----
#pragma unroll
  for (int t = 0; t < 8; ++t)
#pragma unroll
    for (int r = 0; r < 8; ++r) {
      int qrow = qt * 16 + r + 8 * h2;
      AOb[((size_t)(b * SS + qrow) * NH + qh) * DV + t * 16 + n16] =
          (bf16)oacc[t][r];
    }
}

// =====================================================================
extern "C" void kernel_launch(void* const* d_in, const int* in_sizes, int n_in,
                              void* d_out, int out_size, void* d_ws, size_t ws_size,
                              hipStream_t stream)
{
  const float* hidden = (const float*)d_in[0];
  const float* cosb   = (const float*)d_in[1];
  const float* sinb   = (const float*)d_in[2];
  // d_in[3] attention_mask: pure causal, synthesized in-kernel
  const float* Wq   = (const float*)d_in[4];
  const float* Wk   = (const float*)d_in[5];
  const float* Wv   = (const float*)d_in[6];
  const float* Wo   = (const float*)d_in[7];
  const float* sink = (const float*)d_in[8];

  float* out   = (float*)d_out;                       // (B,S,HID)
  float* probs = out + (size_t)BB * SS * HIDN;        // (B,H,S,S)

  bf16* Qb  = (bf16*)d_ws;                            // B*S*NH*DH
  bf16* Kb  = Qb + (size_t)BB * SS * NH * DH;         // B*S*NKV*DH
  bf16* Vb  = Kb + (size_t)BB * SS * NKV * DH;        // B*S*NKV*DV
  bf16* Vt  = Vb + (size_t)BB * SS * NKV * DV;        // B*NKV*DV*S (transposed)
  bf16* AOb = Vt + (size_t)BB * SS * NKV * DV;        // B*S*NH*DV

  const int M = BB * SS;
  dim3 blk(256);

  gemm_bf16_wmma<float, bf16><<<dim3((NH * DH) / 64, M / 128), blk, 0, stream>>>(
      hidden, Wq, Qb, M, NH * DH, HIDN);
  gemm_bf16_wmma<float, bf16><<<dim3((NKV * DH) / 64, M / 128), blk, 0, stream>>>(
      hidden, Wk, Kb, M, NKV * DH, HIDN);
  gemm_bf16_wmma<float, bf16><<<dim3((NKV * DV) / 64, M / 128), blk, 0, stream>>>(
      hidden, Wv, Vb, M, NKV * DV, HIDN);

  {
    int total = BB * SS * (NH + NKV) * (RP / 2);
    rope_kernel<<<(total + 255) / 256, 256, 0, stream>>>(Qb, Kb, cosb, sinb);
  }
  {
    int total = BB * SS * NKV * DV;
    transpose_v<<<(total + 255) / 256, 256, 0, stream>>>(Vb, Vt);
  }

  zero_f32<<<2048, 256, 0, stream>>>(probs, (size_t)BB * NH * SS * SS);

  attn_kernel<<<dim3(SS / 16, NH, BB), 32, 0, stream>>>(
      Qb, Kb, Vt, sink, probs, AOb);

  gemm_bf16_wmma<bf16, float><<<dim3(HIDN / 64, M / 128), blk, 0, stream>>>(
      AOb, Wo, out, M, HIDN, NH * DV);
}
